// Block_4724464025672
// MI455X (gfx1250) — compile-verified, matrix operand order checked by reading, not run on hardware
//
#include <hip/hip_runtime.h>
#include <math.h>

#define kE 384
#define kH 6
#define kD 64
#define kT 256
#define kB 128
#define kM (kB * kT)      // 32768 rows
#define kF (4 * kE)       // 1536

typedef __attribute__((ext_vector_type(16))) _Float16 v16h;
typedef __attribute__((ext_vector_type(8)))  float    v8f;
typedef __attribute__((ext_vector_type(4)))  int      v4i;

// ---------------------------------------------------------------------------
// Async global->LDS staging (CDNA5 GLOBAL_LOAD_ASYNC_TO_LDS_B128, ASYNCcnt)
// ---------------------------------------------------------------------------
#if defined(__AMDGCN__) && __has_builtin(__builtin_amdgcn_global_load_async_to_lds_b128)
#define HAVE_ASYNC_LDS 1
typedef __attribute__((address_space(1))) v4i gv4i;
typedef __attribute__((address_space(3))) v4i lv4i;
#else
#define HAVE_ASYNC_LDS 0
#endif

__device__ __forceinline__ void copy_b128(const _Float16* g, _Float16* l) {
#if HAVE_ASYNC_LDS
    __builtin_amdgcn_global_load_async_to_lds_b128((gv4i*)g, (lv4i*)l, 0, 0);
#else
    *(uint4*)l = *(const uint4*)g;
#endif
}

__device__ __forceinline__ void wait_async_copies() {
#if HAVE_ASYNC_LDS
#if __has_builtin(__builtin_amdgcn_s_wait_asynccnt)
    __builtin_amdgcn_s_wait_asynccnt(0);
#else
    asm volatile("s_wait_asynccnt 0" ::: "memory");
#endif
#endif
}

// ---------------------------------------------------------------------------
// f32 -> f16 cast with transpose: dst[N,K] = src[K,N]^T (weights, amortized)
// ---------------------------------------------------------------------------
__global__ void cast_transpose_kernel(const float* __restrict__ src,
                                      _Float16* __restrict__ dst,
                                      int K, int N) {
    int i = blockIdx.x * blockDim.x + threadIdx.x;
    if (i < K * N) {
        int k = i / N, n = i - k * N;
        dst[(size_t)n * K + k] = (_Float16)src[i];
    }
}

// ---------------------------------------------------------------------------
// LayerNorm: one wave per row of 384; output f16
// ---------------------------------------------------------------------------
__global__ __launch_bounds__(256) void ln_kernel(const float* __restrict__ x,
                                                 const float* __restrict__ g,
                                                 const float* __restrict__ bta,
                                                 _Float16* __restrict__ out) {
    const int wave = threadIdx.x >> 5;
    const int lane = threadIdx.x & 31;
    const int row  = blockIdx.x * 8 + wave;
    const float* xr = x + (size_t)row * kE;

    float vals[12];
    float s = 0.f, ss = 0.f;
#pragma unroll
    for (int i = 0; i < 12; ++i) {
        float v = xr[lane + i * 32];
        vals[i] = v; s += v; ss += v * v;
    }
#pragma unroll
    for (int off = 16; off >= 1; off >>= 1) {
        s  += __shfl_xor(s,  off, 32);
        ss += __shfl_xor(ss, off, 32);
    }
    const float mu   = s * (1.f / kE);
    const float var  = ss * (1.f / kE) - mu * mu;
    const float rstd = rsqrtf(var + 1e-5f);

    _Float16* orow = out + (size_t)row * kE;
#pragma unroll
    for (int i = 0; i < 12; ++i) {
        int c = lane + i * 32;
        orow[c] = (_Float16)((vals[i] - mu) * rstd * g[c] + bta[c]);
    }
}

// ---------------------------------------------------------------------------
// WMMA GEMM: out[M,N] = A[M,K] @ W[K,N] (+bias) (relu?) (+resid)
// WT[N,K] pre-transposed so both operands stage k-contiguous.
// Block: 256 threads = 8 waves. Tile BM=128, BN=64, BK=32, double-buffered:
// async DMA of tile i+1 overlaps WMMA of tile i; one barrier per iteration.
// LDS strides padded (32 -> 40 halves) for conflict-free fragment reads.
// ---------------------------------------------------------------------------
#define LDA 40
template <bool OUT_F16, bool RELU, bool HAS_BIAS, bool HAS_RESID>
__global__ __launch_bounds__(256) void gemm_wmma(
    const _Float16* __restrict__ A, const _Float16* __restrict__ WT,
    const float* __restrict__ bias, const float* __restrict__ resid,
    _Float16* __restrict__ out16, float* __restrict__ out32,
    int M, int N, int K) {
    __shared__ __align__(16) _Float16 As[2][128 * LDA];
    __shared__ __align__(16) _Float16 Bs[2][64 * LDA];

    const int t    = threadIdx.x;
    const int lane = t & 31;
    const int wave = t >> 5;
    const int hf   = lane >> 4;
    const int l16  = lane & 15;
    const int m0   = blockIdx.x * 128;
    const int n0   = blockIdx.y * 64;
    const int mrow = (wave & 3) * 32;
    const int ncol = (wave >> 2) * 32;

    const int ar = t >> 1, ac = (t & 1) * 16;  // A tile: 128x32, 16 f16/thr
    const int br = t >> 2, bc = (t & 3) * 8;   // B tile:  64x32,  8 f16/thr

    const _Float16* gA = &A[(size_t)(m0 + ar) * K + ac];
    const _Float16* gB = &WT[(size_t)(n0 + br) * K + bc];

    auto stage = [&](int bufi, int k0) {
        copy_b128(gA + k0,     &As[bufi][ar * LDA + ac]);
        copy_b128(gA + k0 + 8, &As[bufi][ar * LDA + ac + 8]);
        copy_b128(gB + k0,     &Bs[bufi][br * LDA + bc]);
    };

    v8f cc[2][2] = {};

    stage(0, 0);
    wait_async_copies();
    __syncthreads();

    int buf = 0;
    for (int k0 = 0; k0 < K; k0 += 32) {
        if (k0 + 32 < K) stage(buf ^ 1, k0 + 32);   // DMA next tile now

        v16h a[2], b[2];
#pragma unroll
        for (int v = 0; v < 8; ++v) {
            const int kk = (v >> 2) * 16 + hf * 8 + (v & 3) * 2;
#pragma unroll
            for (int f = 0; f < 2; ++f) {
                a[f][2 * v]     = As[buf][(mrow + f * 16 + l16) * LDA + kk];
                a[f][2 * v + 1] = As[buf][(mrow + f * 16 + l16) * LDA + kk + 1];
                b[f][2 * v]     = Bs[buf][(ncol + f * 16 + l16) * LDA + kk];
                b[f][2 * v + 1] = Bs[buf][(ncol + f * 16 + l16) * LDA + kk + 1];
            }
        }
#pragma unroll
        for (int f = 0; f < 2; ++f)
#pragma unroll
            for (int g = 0; g < 2; ++g)
                cc[f][g] = __builtin_amdgcn_wmma_f32_16x16x32_f16(
                    false, a[f], false, b[g], (short)0, cc[f][g], false, false);

        wait_async_copies();
        __syncthreads();
        buf ^= 1;
    }

#pragma unroll
    for (int f = 0; f < 2; ++f) {
#pragma unroll
        for (int g = 0; g < 2; ++g) {
#pragma unroll
            for (int r = 0; r < 8; ++r) {
                const int m = m0 + mrow + f * 16 + r + hf * 8;
                const int n = n0 + ncol + g * 16 + l16;
                float val = cc[f][g][r];
                if (HAS_BIAS) val += bias[n];
                if (RELU)     val = fmaxf(val, 0.f);
                if (HAS_RESID) val += resid[(size_t)m * N + n];
                if (OUT_F16) out16[(size_t)m * N + n] = (_Float16)val;
                else         out32[(size_t)m * N + n] = val;
            }
        }
    }
}

// ---------------------------------------------------------------------------
// Flash attention: block = (128-query slab, head, batch), 8 waves.
// Wave owns 16 query rows (Q as two A-frags in registers); 32-key K/V tiles
// double-buffered through LDS: K via async-to-LDS b128, V transposed on the
// fly to d-major. Mask/cndmask only on wave-uniform diagonal tiles.
// ---------------------------------------------------------------------------
#define LDK 72   // K tile stride: 64 + 8 halves
#define LDV 40   // V^T / P strides: 32 + 8 halves
__global__ __launch_bounds__(256) void attn_kernel(
    const _Float16* __restrict__ q, const _Float16* __restrict__ k,
    const _Float16* __restrict__ v, _Float16* __restrict__ out, float scale) {
    __shared__ __align__(16) _Float16 Ks[2][32 * LDK];   // [key][d]
    __shared__ __align__(16) _Float16 Vt[2][64 * LDV];   // [d][key]
    __shared__ __align__(16) _Float16 Ps[8 * 16 * LDV];

    const int t    = threadIdx.x;
    const int lane = t & 31;
    const int wave = t >> 5;
    const int hf   = lane >> 4;
    const int l16  = lane & 15;
    const int qt   = blockIdx.x;
    const int h    = blockIdx.y;
    const int b    = blockIdx.z;
    const int qrow0 = qt * 128 + wave * 16;

    // Q fragments (A: 16x64 -> two 16x32 frags) held in registers
    v16h aq0, aq1;
    {
        const size_t qbase = ((size_t)(b * kT + qrow0 + l16) * kH + h) * kD;
#pragma unroll
        for (int vv = 0; vv < 8; ++vv) {
            const int kk = (vv >> 2) * 16 + hf * 8 + (vv & 3) * 2;
            aq0[2 * vv]     = q[qbase + kk];
            aq0[2 * vv + 1] = q[qbase + kk + 1];
            aq1[2 * vv]     = q[qbase + 32 + kk];
            aq1[2 * vv + 1] = q[qbase + 32 + kk + 1];
        }
    }

    float mst[8], lst[8];
#pragma unroll
    for (int r = 0; r < 8; ++r) { mst[r] = -3.0e38f; lst[r] = 0.f; }
    v8f oacc[4] = {};

    const int ir = t >> 3, d8 = (t & 7) * 8;   // K tile: 32 keys x 64 d
    const int vd = t & 63, vkg = t >> 6;       // V^T: thread -> (d, key-group)

    auto stage_kv = [&](int bufi, int kbase) {
        copy_b128(&k[((size_t)(b * kT + kbase + ir) * kH + h) * kD + d8],
                  &Ks[bufi][ir * LDK + d8]);
        __align__(16) _Float16 tmp[8];
        const _Float16* vp = &v[((size_t)(b * kT + kbase + vkg * 8) * kH + h) * kD + vd];
#pragma unroll
        for (int j = 0; j < 8; ++j) tmp[j] = vp[(size_t)j * kE];
        *(uint4*)(&Vt[bufi][vd * LDV + vkg * 8]) = *(const uint4*)tmp;
    };

    const int kmax = qt * 128 + 128;
    stage_kv(0, 0);
    wait_async_copies();
    __syncthreads();

    int buf = 0;
    for (int kb = 0; kb < kmax; kb += 32) {
        if (kb + 32 < kmax) stage_kv(buf ^ 1, kb + 32);  // DMA next tile now

        if (kb <= qrow0 + 15) {   // wave-uniform causal bound
            // S = Q (16x64) @ K^T (64x32) -> two 16x16 f32 tiles
            v8f s[2] = {};
#pragma unroll
            for (int nt = 0; nt < 2; ++nt) {
                v16h b0, b1;
                const int col = nt * 16 + l16;
#pragma unroll
                for (int vv = 0; vv < 8; ++vv) {
                    const int kk = (vv >> 2) * 16 + hf * 8 + (vv & 3) * 2;
                    b0[2 * vv]     = Ks[buf][col * LDK + kk];
                    b0[2 * vv + 1] = Ks[buf][col * LDK + kk + 1];
                    b1[2 * vv]     = Ks[buf][col * LDK + 32 + kk];
                    b1[2 * vv + 1] = Ks[buf][col * LDK + 32 + kk + 1];
                }
                s[nt] = __builtin_amdgcn_wmma_f32_16x16x32_f16(
                    false, aq0, false, b0, (short)0, s[nt], false, false);
                s[nt] = __builtin_amdgcn_wmma_f32_16x16x32_f16(
                    false, aq1, false, b1, (short)0, s[nt], false, false);
            }

            // online softmax; mask only on diagonal tiles (wave-uniform)
            const bool needmask = (kb + 31 > qrow0);
#pragma unroll
            for (int r = 0; r < 8; ++r) {
                const int m = qrow0 + r + hf * 8;
                float s0 = s[0][r] * scale;
                float s1 = s[1][r] * scale;
                if (needmask) {
                    if (kb + l16 > m)      s0 = -3.0e38f;
                    if (kb + 16 + l16 > m) s1 = -3.0e38f;
                }
                float tmax = fmaxf(s0, s1);
#pragma unroll
                for (int off = 8; off >= 1; off >>= 1)
                    tmax = fmaxf(tmax, __shfl_xor(tmax, off, 32));
                const float nm  = fmaxf(mst[r], tmax);
                const float fsc = __expf(mst[r] - nm);
                const float p0  = __expf(s0 - nm);
                const float p1  = __expf(s1 - nm);
                float rs = p0 + p1;
#pragma unroll
                for (int off = 8; off >= 1; off >>= 1)
                    rs += __shfl_xor(rs, off, 32);
                lst[r] = lst[r] * fsc + rs;
                mst[r] = nm;
                oacc[0][r] *= fsc; oacc[1][r] *= fsc;
                oacc[2][r] *= fsc; oacc[3][r] *= fsc;
                Ps[wave * 16 * LDV + (r + hf * 8) * LDV + l16]      = (_Float16)p0;
                Ps[wave * 16 * LDV + (r + hf * 8) * LDV + 16 + l16] = (_Float16)p1;
            }

            // P (16x32) as A-fragment (contiguous b128 reads, wave-private)
            v16h pa;
#pragma unroll
            for (int vv = 0; vv < 8; ++vv) {
                const int kk = (vv >> 2) * 16 + hf * 8 + (vv & 3) * 2;
                pa[2 * vv]     = Ps[wave * 16 * LDV + l16 * LDV + kk];
                pa[2 * vv + 1] = Ps[wave * 16 * LDV + l16 * LDV + kk + 1];
            }
            // O (16x64) += P (16x32) @ V (32x64)
#pragma unroll
            for (int dt = 0; dt < 4; ++dt) {
                v16h bv;
                const int col = dt * 16 + l16;
#pragma unroll
                for (int vv = 0; vv < 8; ++vv) {
                    const int kk = (vv >> 2) * 16 + hf * 8 + (vv & 3) * 2;
                    bv[2 * vv]     = Vt[buf][col * LDV + kk];
                    bv[2 * vv + 1] = Vt[buf][col * LDV + kk + 1];
                }
                oacc[dt] = __builtin_amdgcn_wmma_f32_16x16x32_f16(
                    false, pa, false, bv, (short)0, oacc[dt], false, false);
            }
        }

        wait_async_copies();
        __syncthreads();
        buf ^= 1;
    }

#pragma unroll
    for (int r = 0; r < 8; ++r) {
        const float inv = 1.0f / lst[r];
        const int m = qrow0 + r + hf * 8;
        const size_t obase = ((size_t)(b * kT + m) * kH + h) * kD;
#pragma unroll
        for (int dt = 0; dt < 4; ++dt)
            out[obase + dt * 16 + l16] = (_Float16)(oacc[dt][r] * inv);
    }
}

// ---------------------------------------------------------------------------
extern "C" void kernel_launch(void* const* d_in, const int* in_sizes, int n_in,
                              void* d_out, int out_size, void* d_ws,
                              size_t ws_size, hipStream_t stream) {
    const float* x     = (const float*)d_in[0];
    const float* wq    = (const float*)d_in[1];
    const float* wk    = (const float*)d_in[2];
    const float* wv    = (const float*)d_in[3];
    const float* wproj = (const float*)d_in[4];
    const float* bproj = (const float*)d_in[5];
    const float* w1    = (const float*)d_in[6];
    const float* b1    = (const float*)d_in[7];
    const float* w2    = (const float*)d_in[8];
    const float* b2    = (const float*)d_in[9];
    const float* g1    = (const float*)d_in[10];
    const float* bt1   = (const float*)d_in[11];
    const float* g2    = (const float*)d_in[12];
    const float* bt2   = (const float*)d_in[13];
    float* out = (float*)d_out;

    // f16 workspace layout (transposed weights: [N,K])
    _Float16* wsp = (_Float16*)d_ws;
    size_t off = 0;
    _Float16* wqT = wsp + off; off += (size_t)kE * kE;
    _Float16* wkT = wsp + off; off += (size_t)kE * kE;
    _Float16* wvT = wsp + off; off += (size_t)kE * kE;
    _Float16* wpT = wsp + off; off += (size_t)kE * kE;
    _Float16* w1T = wsp + off; off += (size_t)kE * kF;
    _Float16* w2T = wsp + off; off += (size_t)kF * kE;
    _Float16* xn1 = wsp + off; off += (size_t)kM * kE;
    _Float16* qb  = wsp + off; off += (size_t)kM * kE;
    _Float16* kb  = wsp + off; off += (size_t)kM * kE;
    _Float16* vb  = wsp + off; off += (size_t)kM * kE;
    _Float16* ab  = wsp + off; off += (size_t)kM * kE;
    _Float16* xn2 = wsp + off; off += (size_t)kM * kE;
    _Float16* h1  = wsp + off; off += (size_t)kM * kF;

    const int nEE = kE * kE, nEF = kE * kF;
    cast_transpose_kernel<<<(nEE + 255) / 256, 256, 0, stream>>>(wq, wqT, kE, kE);
    cast_transpose_kernel<<<(nEE + 255) / 256, 256, 0, stream>>>(wk, wkT, kE, kE);
    cast_transpose_kernel<<<(nEE + 255) / 256, 256, 0, stream>>>(wv, wvT, kE, kE);
    cast_transpose_kernel<<<(nEE + 255) / 256, 256, 0, stream>>>(wproj, wpT, kE, kE);
    cast_transpose_kernel<<<(nEF + 255) / 256, 256, 0, stream>>>(w1, w1T, kE, kF);
    cast_transpose_kernel<<<(nEF + 255) / 256, 256, 0, stream>>>(w2, w2T, kF, kE);

    // LN1
    ln_kernel<<<kM / 8, 256, 0, stream>>>(x, g1, bt1, xn1);

    // QKV projections
    dim3 gEE(kM / 128, kE / 64);
    gemm_wmma<true, false, false, false><<<gEE, 256, 0, stream>>>(
        xn1, wqT, nullptr, nullptr, qb, nullptr, kM, kE, kE);
    gemm_wmma<true, false, false, false><<<gEE, 256, 0, stream>>>(
        xn1, wkT, nullptr, nullptr, kb, nullptr, kM, kE, kE);
    gemm_wmma<true, false, false, false><<<gEE, 256, 0, stream>>>(
        xn1, wvT, nullptr, nullptr, vb, nullptr, kM, kE, kE);

    // attention (reference scales by full embed dim: 384^-0.5)
    attn_kernel<<<dim3(kT / 128, kH, kB), 256, 0, stream>>>(
        qb, kb, vb, ab, 1.0f / sqrtf((float)kE));

    // out-proj + bias + residual(x) -> x2 (f32 in d_out)
    gemm_wmma<false, false, true, true><<<gEE, 256, 0, stream>>>(
        ab, wpT, bproj, x, nullptr, out, kM, kE, kE);

    // LN2
    ln_kernel<<<kM / 8, 256, 0, stream>>>(out, g2, bt2, xn2);

    // MLP: relu(xn2 @ w1 + b1) -> h1 (f16)
    dim3 gEF(kM / 128, kF / 64);
    gemm_wmma<true, true, true, false><<<gEF, 256, 0, stream>>>(
        xn2, w1T, b1, nullptr, h1, nullptr, kM, kF, kE);

    // h1 @ w2 + b2 + residual(x2) -> out (in-place f32)
    gemm_wmma<false, false, true, true><<<gEE, 256, 0, stream>>>(
        h1, w2T, b2, out, nullptr, out, kM, kE, kF);
}
